// MQSPNDetCRF_54735063220593
// MI455X (gfx1250) — matrix operand order repaired and checked by reading, not compiled, not run on hardware
//
#include <hip/hip_runtime.h>
#include <hip/hip_fp16.h>

typedef __attribute__((ext_vector_type(16))) _Float16 v16h;
typedef __attribute__((ext_vector_type(8)))  float    v8f;
typedef __attribute__((ext_vector_type(4)))  int      i4v;

#define BB 1024
#define TT 512
#define KK 64
#define CL 16            // backtrack chunk length
#define NC (TT / CL)     // 32 chunks

__device__ __forceinline__ float nexp(float x) {
    return __builtin_amdgcn_exp2f(x * 1.4426950408889634f);   // v_exp_f32
}
__device__ __forceinline__ float nlog(float x) {
    return __builtin_amdgcn_logf(x) * 0.6931471805599453f;    // v_log_f32
}

// CDNA5 async global->LDS copy path (ASYNCcnt), guarded: falls back to
// synchronous VGPR staging when the builtins are absent on this toolchain.
#if __has_builtin(__builtin_amdgcn_global_load_async_to_lds_b128) && \
    __has_builtin(__builtin_amdgcn_s_wait_asynccnt)
#define ASYNC_LDS 1
typedef __attribute__((address_space(1))) i4v g_i4;
typedef __attribute__((address_space(3))) i4v l_i4;
__device__ __forceinline__ void async_b128(const void* g, void* l) {
    __builtin_amdgcn_global_load_async_to_lds_b128((g_i4*)g, (l_i4*)l, 0, 0);
}
#else
#define ASYNC_LDS 0
#endif

// ---------------------------------------------------------------------------
// Kernel 1: log-partition via WMMA.
//   alpha'[j] = log( sum_i exp(alpha[i]-m) * E[i][j] ) + m + em[t][j]
// Block = 16 batch rows; 4 waves; wave nt computes N-tile nt via
// v_wmma_f32_16x16x32_f16 (K=64 as 2 chunks). Row-max by width-8 shuffles.
// Emissions double-buffered in LDS via GLOBAL_LOAD_ASYNC_TO_LDS when available.
// ---------------------------------------------------------------------------
__global__ __launch_bounds__(128) void crf_forward_wmma(
    const float* __restrict__ em, const int* __restrict__ mask,
    const float* __restrict__ trans, const float* __restrict__ start,
    const float* __restrict__ endt, float* __restrict__ logZ)
{
    __shared__ __align__(16) float    alpha_s[16][64];
    __shared__ __align__(16) float    em_s[2][16][64];  // double buffer
    __shared__ __align__(16) _Float16 pbuf[16][64];     // exp(alpha-m), A-source
    __shared__ __align__(16) _Float16 etrT[64][64];     // etrT[n][k] = exp(Tr[k][n])
    __shared__ float m_s[16];
    __shared__ int   msk_s[16];

    const int tid  = threadIdx.x;       // 0..127
    const int lane = tid & 31;
    const int nt   = tid >> 5;          // wave id = N-tile
    const int half = lane >> 4;
    const int l16  = lane & 15;
    const int row  = tid >> 3;          // 0..15 : 8 lanes (one shuffle group) per row
    const int colb = (tid & 7) * 8;     // 8 cols per lane
    const int b0   = blockIdx.x * 16;

    // E^T = exp(Tr)^T into LDS (once)
    for (int e = tid; e < 64 * 64; e += 128) {
        int i = e >> 6, n = e & 63;
        etrT[n][i] = (_Float16)nexp(trans[e]);
    }
    // alpha0 = start + em[:,0]
#pragma unroll
    for (int j = 0; j < 8; ++j) {
        int col = colb + j;
        alpha_s[row][col] = start[col] + em[((size_t)(b0 + row) * TT) * KK + col];
    }
#if ASYNC_LDS
    {   // prologue: stage em[t=1] into buffer 1 (32B per lane = 2 x b128)
        const float* g = &em[((size_t)(b0 + row) * TT + 1) * KK + colb];
        async_b128(g,     &em_s[1][row][colb]);
        async_b128(g + 4, &em_s[1][row][colb + 4]);
    }
#endif
    __syncthreads();

    // Resident B-fragments: contiguous k order (k = 16h+s) -> N = nt*16+l16,
    // half selects K 0-15 / 16-31; 32 contiguous bytes of etrT[N][...].
    v16h bfrag[2];
    {
        const _Float16* bsrc = &etrT[nt * 16 + l16][0];
        for (int c = 0; c < 2; ++c) {
            const float4* s = (const float4*)(bsrc + c * 32 + half * 16);
            ((float4*)&bfrag[c])[0] = s[0];
            ((float4*)&bfrag[c])[1] = s[1];
        }
    }

    for (int tt = 1; tt < TT; ++tt) {
        const int cur = tt & 1;
#if ASYNC_LDS
        if (tt + 1 < TT) {   // issue next step's emissions while we compute
            const float* g = &em[((size_t)(b0 + row) * TT + tt + 1) * KK + colb];
            async_b128(g,     &em_s[cur ^ 1][row][colb]);
            async_b128(g + 4, &em_s[cur ^ 1][row][colb + 4]);
        }
#endif
        // --- fused phase: row max via shuffles, p = exp(a-m) ---
        float av[8];
        float pmax = -3.0e38f;
#pragma unroll
        for (int j = 0; j < 8; ++j) {
            av[j] = alpha_s[row][colb + j];
            pmax  = fmaxf(pmax, av[j]);
        }
#if !ASYNC_LDS
        float ev[8];
        int ttn = (tt + 1 < TT) ? tt + 1 : tt;
        __builtin_prefetch(&em[((size_t)(b0 + row) * TT + ttn) * KK + colb], 0, 1);
#pragma unroll
        for (int j = 0; j < 8; ++j)
            ev[j] = em[((size_t)(b0 + row) * TT + tt) * KK + colb + j];
#endif
        pmax = fmaxf(pmax, __shfl_xor(pmax, 1, 8));
        pmax = fmaxf(pmax, __shfl_xor(pmax, 2, 8));
        pmax = fmaxf(pmax, __shfl_xor(pmax, 4, 8));
        if ((tid & 7) == 0) {
            m_s[row]   = pmax;
            msk_s[row] = mask[(b0 + row) * TT + tt];
        }
#pragma unroll
        for (int j = 0; j < 8; ++j) {
#if !ASYNC_LDS
            em_s[cur][row][colb + j] = ev[j];
#endif
            pbuf[row][colb + j] = (_Float16)nexp(av[j] - pmax);
        }
#if ASYNC_LDS
        // current step's tile complete (in-order): allow the just-issued batch
        if (tt + 1 < TT) __builtin_amdgcn_s_wait_asynccnt(2);
        else             __builtin_amdgcn_s_wait_asynccnt(0);
#endif
        __syncthreads();

        // --- WMMA: s = p * E (A from pbuf, contiguous k order matching B) ---
        v16h afrag[2];
        {
            const _Float16* ap = &pbuf[l16][0];
            for (int c = 0; c < 2; ++c) {
                ((float4*)&afrag[c])[0] = *(const float4*)(ap + c * 32 + half * 16);
                ((float4*)&afrag[c])[1] = *(const float4*)(ap + c * 32 + half * 16 + 8);
            }
        }
        v8f acc = {};
        acc = __builtin_amdgcn_wmma_f32_16x16x32_f16(false, afrag[0], false, bfrag[0],
                                                     (short)0, acc, false, false);
        acc = __builtin_amdgcn_wmma_f32_16x16x32_f16(false, afrag[1], false, bfrag[1],
                                                     (short)0, acc, false, false);
        // --- writeback: alpha' = log(s) + m + em, masked (C layout) ---
        {
            int N = nt * 16 + l16;
#pragma unroll
            for (int r = 0; r < 8; ++r) {
                int M = r + 8 * half;
                float an  = nlog(acc[r]) + m_s[M] + em_s[cur][M][N];
                float old = alpha_s[M][N];
                alpha_s[M][N] = msk_s[M] ? an : old;
            }
        }
        __syncthreads();
    }

    // logZ = logsumexp(alpha + end), row groups of 8 lanes + shuffles
    {
        float se[8];
        float pmax = -3.0e38f;
#pragma unroll
        for (int j = 0; j < 8; ++j) {
            se[j] = alpha_s[row][colb + j] + endt[colb + j];
            pmax  = fmaxf(pmax, se[j]);
        }
        pmax = fmaxf(pmax, __shfl_xor(pmax, 1, 8));
        pmax = fmaxf(pmax, __shfl_xor(pmax, 2, 8));
        pmax = fmaxf(pmax, __shfl_xor(pmax, 4, 8));
        float psum = 0.f;
#pragma unroll
        for (int j = 0; j < 8; ++j) psum += nexp(se[j] - pmax);
        psum += __shfl_xor(psum, 1, 8);
        psum += __shfl_xor(psum, 2, 8);
        psum += __shfl_xor(psum, 4, 8);
        if ((tid & 7) == 0) logZ[b0 + row] = pmax + nlog(psum);
    }
}

// ---------------------------------------------------------------------------
// Kernel 2: Viterbi forward (exact f32 max-plus). Block = 4 rows x 64 tags.
// Transition column in 64 VGPRs; v-vector broadcast from LDS; bp as bytes.
// ---------------------------------------------------------------------------
__global__ __launch_bounds__(256) void crf_viterbi_fwd(
    const float* __restrict__ em, const int* __restrict__ mask,
    const float* __restrict__ trans, const float* __restrict__ start,
    const float* __restrict__ endt,
    unsigned char* __restrict__ bp_ws, int* __restrict__ last_ws)
{
    __shared__ float v_s[4][64];
    const int tid  = threadIdx.x;
    const int j    = tid & 63;
    const int brow = tid >> 6;
    const int b    = blockIdx.x * 4 + brow;

    float trc[64];
#pragma unroll
    for (int i = 0; i < 64; ++i) trc[i] = trans[i * KK + j];   // Tr[i][j]

    v_s[brow][j] = start[j] + em[((size_t)b * TT) * KK + j];
    __syncthreads();

    for (int tt = 1; tt < TT; ++tt) {
        float e_t = em[((size_t)b * TT + tt) * KK + j];
        int   mk  = mask[b * TT + tt];
        float best = -3.0e38f;
        int   bi   = 0;
#pragma unroll
        for (int i = 0; i < 64; ++i) {
            float s = v_s[brow][i] + trc[i];
            if (s > best) { best = s; bi = i; }
        }
        float nv = best + e_t;
        if (!mk) { nv = v_s[brow][j]; bi = j; }   // masked: keep v, identity bp
        bp_ws[((size_t)tt * BB + b) * KK + j] = (unsigned char)bi;
        __syncthreads();
        v_s[brow][j] = nv;
        __syncthreads();
    }

    if (j == 0) {
        float best = -3.0e38f;
        int   bi   = 0;
        for (int i = 0; i < 64; ++i) {
            float s = v_s[brow][i] + endt[i];
            if (s > best) { best = s; bi = i; }
        }
        last_ws[b] = bi;
    }
}

// ---------------------------------------------------------------------------
// Kernel 3: gold score per batch row.
// ---------------------------------------------------------------------------
__global__ __launch_bounds__(128) void crf_gold(
    const float* __restrict__ em, const int* __restrict__ tags,
    const int* __restrict__ mask, const float* __restrict__ trans,
    const float* __restrict__ start, const float* __restrict__ endt,
    float* __restrict__ score_ws)
{
    __shared__ float s_em[128], s_tr[128];
    __shared__ int   s_m[128];
    const int b  = blockIdx.x;
    const int t0 = threadIdx.x;
    float em_acc = 0.f, tr_acc = 0.f;
    int msum = 0;
    for (int t = t0; t < TT; t += 128) {
        int tag = tags[b * TT + t];
        int mk  = mask[b * TT + t];
        em_acc += mk ? em[((size_t)b * TT + t) * KK + tag] : 0.f;
        msum   += mk ? 1 : 0;
        if (t > 0) {
            int ptag = tags[b * TT + t - 1];
            tr_acc += mk ? trans[ptag * KK + tag] : 0.f;
        }
    }
    s_em[t0] = em_acc; s_tr[t0] = tr_acc; s_m[t0] = msum;
    __syncthreads();
    for (int off = 64; off > 0; off >>= 1) {
        if (t0 < off) {
            s_em[t0] += s_em[t0 + off];
            s_tr[t0] += s_tr[t0 + off];
            s_m[t0]  += s_m[t0 + off];
        }
        __syncthreads();
    }
    if (t0 == 0) {
        int last = s_m[0] - 1;
        score_ws[b] = s_em[0] + s_tr[0] + start[tags[b * TT]] + endt[tags[b * TT + last]];
    }
}

// ---------------------------------------------------------------------------
// Kernel 4a: per-chunk backpointer composition (parallel over all exit tags).
// map[c][b][j] = tag at position c*CL, given tag j at position min((c+1)*CL, TT-1)
// ---------------------------------------------------------------------------
__global__ __launch_bounds__(64) void crf_bt_chunkmap(
    const unsigned char* __restrict__ bp_ws, unsigned char* __restrict__ map_ws)
{
    __shared__ int rowbuf_i[16];
    unsigned char* rowbuf = (unsigned char*)rowbuf_i;
    const int blk = blockIdx.x;        // 0 .. B*NC-1
    const int b   = blk >> 5;          // NC == 32
    const int c   = blk & (NC - 1);
    const int j   = threadIdx.x;
    const int t_hi = min((c + 1) * CL, TT - 1);
    const int t_lo = c * CL + 1;
    int tag = j;
    for (int t = t_hi; t >= t_lo; --t) {
        if (j < 16)
            rowbuf_i[j] = ((const int*)(bp_ws + ((size_t)t * BB + b) * KK))[j];
        __syncthreads();
        tag = rowbuf[tag];
        __syncthreads();
    }
    map_ws[((size_t)c * BB + b) * KK + j] = (unsigned char)tag;
}

// ---------------------------------------------------------------------------
// Kernel 4b: compose chunk maps per sequence (serial depth NC=32); also nll.
// btag[c][b] = tag at position t_hi(c) entering chunk c.
// ---------------------------------------------------------------------------
__global__ __launch_bounds__(256) void crf_bt_compose(
    const float* __restrict__ logZ, const float* __restrict__ score,
    const unsigned char* __restrict__ map_ws, const int* __restrict__ last_ws,
    unsigned char* __restrict__ btag_ws, float* __restrict__ out)
{
    int b = blockIdx.x * blockDim.x + threadIdx.x;
    if (b >= BB) return;
    out[b] = logZ[b] - score[b];
    int tag = last_ws[b];                       // tag at position TT-1
    for (int c = NC - 1; c >= 0; --c) {
        btag_ws[(size_t)c * BB + b] = (unsigned char)tag;          // tag at t_hi(c)
        tag = map_ws[((size_t)c * BB + b) * KK + tag];             // tag at c*CL
    }
}

// ---------------------------------------------------------------------------
// Kernel 4c: re-walk each chunk in parallel, writing the path as float.
// ---------------------------------------------------------------------------
__global__ __launch_bounds__(64) void crf_bt_fill(
    const unsigned char* __restrict__ bp_ws, const unsigned char* __restrict__ btag_ws,
    float* __restrict__ out)
{
    __shared__ int rowbuf_i[16];
    unsigned char* rowbuf = (unsigned char*)rowbuf_i;
    const int blk = blockIdx.x;
    const int b   = blk >> 5;
    const int c   = blk & (NC - 1);
    const int j   = threadIdx.x;
    const int t_hi = min((c + 1) * CL, TT - 1);
    const int t_lo = c * CL + 1;
    int tag = btag_ws[(size_t)c * BB + b];      // tag at position t_hi
    float* path = out + BB + (size_t)b * TT;
    if (j == 0 && c == NC - 1) path[TT - 1] = (float)tag;
    for (int t = t_hi; t >= t_lo; --t) {
        if (j < 16)
            rowbuf_i[j] = ((const int*)(bp_ws + ((size_t)t * BB + b) * KK))[j];
        __syncthreads();
        tag = rowbuf[tag];
        if (j == 0) path[t - 1] = (float)tag;
        __syncthreads();
    }
}

// ---------------------------------------------------------------------------
extern "C" void kernel_launch(void* const* d_in, const int* in_sizes, int n_in,
                              void* d_out, int out_size, void* d_ws, size_t ws_size,
                              hipStream_t stream)
{
    (void)in_sizes; (void)n_in; (void)out_size; (void)ws_size;
    const float* emissions = (const float*)d_in[0];   // [B,T,K] f32
    const int*   tags      = (const int*)d_in[1];     // [B,T]
    const int*   mask      = (const int*)d_in[2];     // [B,T]
    const float* trans     = (const float*)d_in[3];   // [K,K]
    const float* startt    = (const float*)d_in[4];   // [K]
    const float* endt      = (const float*)d_in[5];   // [K]
    float*       out       = (float*)d_out;           // [B] nll + [B,T] path

    // workspace layout
    unsigned char* ws   = (unsigned char*)d_ws;
    float*         ws_logZ  = (float*)(ws);                   // 4 KB
    float*         ws_score = (float*)(ws + 4096);            // 4 KB
    int*           ws_last  = (int*)(ws + 8192);              // 4 KB
    unsigned char* ws_btag  = ws + 12288;                     // NC*B   = 32 KB
    unsigned char* ws_map   = ws + 65536;                     // NC*B*K = 2 MB
    unsigned char* ws_bp    = ws + (4u << 20);                // T*B*K  = 33.5 MB

    crf_forward_wmma<<<BB / 16, 128, 0, stream>>>(emissions, mask, trans, startt, endt, ws_logZ);
    crf_viterbi_fwd <<<BB / 4, 256, 0, stream>>>(emissions, mask, trans, startt, endt, ws_bp, ws_last);
    crf_gold        <<<BB, 128, 0, stream>>>(emissions, tags, mask, trans, startt, endt, ws_score);
    crf_bt_chunkmap <<<BB * NC, 64, 0, stream>>>(ws_bp, ws_map);
    crf_bt_compose  <<<BB / 256, 256, 0, stream>>>(ws_logZ, ws_score, ws_map, ws_last, ws_btag, out);
    crf_bt_fill     <<<BB * NC, 64, 0, stream>>>(ws_bp, ws_btag, out);
}
